// MCFA_19774029431781
// MI455X (gfx1250) — matrix operand order repaired and checked by reading, not compiled, not run on hardware
//
#include <hip/hip_runtime.h>
#include <hip/hip_bf16.h>
#include <math.h>

// ---------------------------------------------------------------------------
// MCFA axial-attention block for gfx1250 (MI455X).
// bf16 NHWC intermediates; v_wmma_f32_16x16x32_bf16 for the three pointwise
// GEMMs (dproj 64->128, v 64->128, proj 320->64) with double-buffered
// GLOBAL_LOAD_ASYNC_TO_LDS_B128 staging; VALU for LN / depthwise convs /
// pooling / tiny 16x16 axial attention.
// ---------------------------------------------------------------------------

typedef __bf16 bf16_t;
typedef __bf16 v8bf  __attribute__((ext_vector_type(8)));
typedef __bf16 v16bf __attribute__((ext_vector_type(16)));
typedef float  v8f   __attribute__((ext_vector_type(8)));

#define BB   2
#define HH   512
#define WW   512
#define HWP  (HH*WW)        // 262144 = 2^18
#define PIX  (BB*HWP)       // 524288

__device__ __forceinline__ float gelu_f(float x) {
  return 0.5f * x * (1.0f + erff(x * 0.70710678118654752440f));
}
__device__ __forceinline__ float  fb(bf16_t v) { return (float)v; }
__device__ __forceinline__ bf16_t tb(float v)  { return (bf16_t)v; }

union U16 { v16bf v; v8bf h[2]; };

// Async global->LDS 16B copy (CDNA5 GLOBAL_LOAD_ASYNC_TO_LDS_B128, ASYNCcnt).
// ldsoff: wave-relative LDS byte address (low 32 bits of generic pointer).
__device__ __forceinline__ void async_g2l_b128(unsigned ldsoff, const void* gaddr) {
  asm volatile("global_load_async_to_lds_b128 %0, %1, off"
               :: "v"(ldsoff), "v"((unsigned long long)(size_t)gaddr)
               : "memory");
}
__device__ __forceinline__ void wait_async0() {
  asm volatile("s_wait_asynccnt 0x0" ::: "memory");
}
__device__ __forceinline__ unsigned lds_off(const void* p) {
  return (unsigned)(size_t)p;   // addr[31:0] of generic LDS address = LDS offset
}

// --------------------------- weight fp32 -> bf16 ---------------------------
__global__ __launch_bounds__(256) void wconv_kernel(
    const float* __restrict__ w0, const float* __restrict__ w1,
    const float* __restrict__ w2,
    bf16_t* __restrict__ o0, bf16_t* __restrict__ o1, bf16_t* __restrict__ o2) {
  int i = blockIdx.x * 256 + threadIdx.x;
  if (i < 128*64) o0[i] = tb(w0[i]);      // dproj_pw_w (128,64)
  if (i < 128*64) o1[i] = tb(w1[i]);      // v_pw_w    (128,64)
  if (i < 64*320) o2[i] = tb(w2[i]);      // proj_pw_w (64,320)
}

// ------------------- x: NCHW f32 -> NHWC bf16 (64 chans) -------------------
__global__ __launch_bounds__(256) void nchw2nhwc64(const float* __restrict__ in,
                                                   bf16_t* __restrict__ out) {
  __shared__ bf16_t tile[256 * 64];
  int t = threadIdx.x;
  int p = blockIdx.x * 256 + t;
  int b = p >> 18, yx = p & (HWP - 1);
  size_t base = (size_t)b * 64 * HWP + yx;
  for (int c = 0; c < 64; ++c) tile[t * 64 + c] = tb(in[base + (size_t)c * HWP]);
  __syncthreads();
  bf16_t* dst = out + (size_t)blockIdx.x * 256 * 64;
  #pragma unroll
  for (int j = 0; j < 8; ++j) {
    int idx = (j * 256 + t) * 8;
    *(v8bf*)(dst + idx) = *(v8bf*)(tile + idx);
  }
}

// ----------------- LayerNorm over channels -> NHWC bf16 --------------------
__global__ __launch_bounds__(256) void ln_kernel(const float* __restrict__ in,
                                                 const float* __restrict__ lnw,
                                                 const float* __restrict__ lnb,
                                                 bf16_t* __restrict__ out) {
  __shared__ bf16_t tile[256 * 64];
  int t = threadIdx.x;
  int p = blockIdx.x * 256 + t;
  int b = p >> 18, yx = p & (HWP - 1);
  size_t base = (size_t)b * 64 * HWP + yx;
  float s = 0.f, ss = 0.f;
  for (int c = 0; c < 64; ++c) {
    float v = in[base + (size_t)c * HWP];
    s += v; ss += v * v;
  }
  float mu  = s * (1.f / 64.f);
  float var = ss * (1.f / 64.f) - mu * mu;
  float rs  = rsqrtf(var + 1e-5f);
  for (int c = 0; c < 64; ++c) {
    float v = in[base + (size_t)c * HWP];
    tile[t * 64 + c] = tb((v - mu) * rs * lnw[c] + lnb[c]);
  }
  __syncthreads();
  bf16_t* dst = out + (size_t)blockIdx.x * 256 * 64;
  #pragma unroll
  for (int j = 0; j < 8; ++j) {
    int idx = (j * 256 + t) * 8;
    *(v8bf*)(dst + idx) = *(v8bf*)(tile + idx);
  }
}

// ------------- depthwise/grouped 3x3 conv + BN + GELU (NHWC) ---------------
// out channel n uses input channel n>>SHIFT. in stride = ins, offset inoff.
template<int OUTC, int SHIFT>
__global__ __launch_bounds__(256) void dw3x3_bn_gelu(
    const bf16_t* __restrict__ in, int ins, int inoff,
    const float* __restrict__ w, const float* __restrict__ bns,
    const float* __restrict__ bnt, bf16_t* __restrict__ out) {
  int bid  = blockIdx.x;
  int xblk = bid & 15, y = (bid >> 4) & 511, b = bid >> 13;
  int t = threadIdx.x;
  int n = t % OUTC;
  int po = t / OUTC;
  const int PPT = 256 / OUTC;
  float w9[9];
  #pragma unroll
  for (int k = 0; k < 9; ++k) w9[k] = w[n * 9 + k];
  float sc = bns[n], sh = bnt[n];
  int cin = n >> SHIFT;
  for (int xi = po; xi < 32; xi += PPT) {
    int x = xblk * 32 + xi;
    float acc = 0.f;
    #pragma unroll
    for (int ky = 0; ky < 3; ++ky) {
      int yy = y + ky - 1;
      if (yy < 0 || yy >= HH) continue;
      #pragma unroll
      for (int kx = 0; kx < 3; ++kx) {
        int xx = x + kx - 1;
        if (xx < 0 || xx >= WW) continue;
        size_t pin = (size_t)(b * HH + yy) * WW + xx;
        acc += fb(in[pin * ins + inoff + cin]) * w9[ky * 3 + kx];
      }
    }
    float v = gelu_f(acc * sc + sh);
    size_t pp = (size_t)(b * HH + y) * WW + x;
    out[pp * OUTC + n] = tb(v);
  }
}

// ----------------------- pointwise GEMM via WMMA ---------------------------
// OUT(pix, CO) = IN(pix, CI) x W(CO, CI)^T.  Block covers 128 pixels x CO.
// B-tile (128px x 32ch) staged via double-buffered async global->LDS copies.
// MODE 0: bias+GELU -> bf16 NHWC ; MODE 1: BN scale/shift -> bf16 NHWC ;
// MODE 2: bias+GELU -> f32 NCHW.
template<int CI, int CO, int INS, int MODE>
__global__ __launch_bounds__(256) void gemm_pw(
    const bf16_t* __restrict__ in, int inoff,
    const bf16_t* __restrict__ W,
    const float* __restrict__ p0, const float* __restrict__ p1,
    bf16_t* __restrict__ oh, float* __restrict__ of) {
  __shared__ bf16_t ldsB[2][128 * 32];
  const int t    = threadIdx.x;
  const int lane = t & 31;
  const int wv   = t >> 5;
  const int pix0 = blockIdx.x * 128;
  constexpr int KI   = CI / 32;
  constexpr int MT   = CO / 16;              // 8 or 4 M-tiles
  constexpr int NTPW = (MT == 8) ? 8 : 4;    // N-tiles per wave
  const int mt  = wv % MT;
  const int ntb = (wv / MT) * NTPW;
  const int lhi = (lane >= 16) ? 1 : 0;
  const int l15 = lane & 15;

  v8f acc[NTPW] = {};

  // per-thread staging slice: 32B of the 128px x 32ch tile
  const int spix = t >> 1;
  const int sch  = (t & 1) * 16;
  const bf16_t* gbase = in + (size_t)(pix0 + spix) * INS + inoff + sch;
  const unsigned loff0 = lds_off(&ldsB[0][spix * 32 + sch]);
  const unsigned loff1 = lds_off(&ldsB[1][spix * 32 + sch]);

  // prologue: async-fill buffer 0 with K-slice 0
  async_g2l_b128(loff0,      gbase);
  async_g2l_b128(loff0 + 16, gbase + 8);

  #pragma unroll 1
  for (int ki = 0; ki < KI; ++ki) {
    wait_async0();        // this wave's fill(ki) landed in LDS
    __syncthreads();      // all waves' fill(ki) landed; readers of fill(ki+1)'s
                          // target buffer (from iter ki-1) are done
    if (ki + 1 < KI) {    // async-fill the other buffer with K-slice ki+1
      unsigned lo = ((ki + 1) & 1) ? loff1 : loff0;
      const bf16_t* g = gbase + (ki + 1) * 32;
      async_g2l_b128(lo,      g);
      async_g2l_b128(lo + 16, g + 8);
    }
    // A-matrix (weights) gather: lane=M, 16 consecutive K
    U16 a;
    {
      int M  = mt * 16 + l15;
      int k0 = ki * 32 + lhi * 16;
      const bf16_t* ar = W + (size_t)M * CI + k0;
      a.h[0] = *(const v8bf*)ar;
      a.h[1] = *(const v8bf*)(ar + 8);
    }
    const bf16_t* bufp = &ldsB[ki & 1][0];
    #pragma unroll
    for (int i = 0; i < NTPW; ++i) {
      int nt = ntb + i;
      // B-matrix gather from LDS: lane=N(pixel), 16 consecutive K
      U16 bm;
      const bf16_t* br = bufp + (nt * 16 + l15) * 32 + lhi * 16;
      bm.h[0] = *(const v8bf*)br;
      bm.h[1] = *(const v8bf*)(br + 8);
      acc[i] = __builtin_amdgcn_wmma_f32_16x16x32_bf16(
          false, a.v, false, bm.v, (short)0, acc[i], false, false);
    }
  }
  // epilogue: C/D layout lane=N(pixel), VGPR r -> M = r + 8*(lane>=16)
  #pragma unroll
  for (int i = 0; i < NTPW; ++i) {
    int nt    = ntb + i;
    int pixel = pix0 + nt * 16 + l15;
    int chb   = mt * 16 + lhi * 8;
    if (MODE == 2) {
      int b = pixel >> 18, yx = pixel & (HWP - 1);
      #pragma unroll
      for (int r = 0; r < 8; ++r) {
        float v = gelu_f(acc[i][r] + p0[chb + r]);
        of[((size_t)b * CO + (chb + r)) * HWP + yx] = v;
      }
    } else {
      v8bf pk;
      #pragma unroll
      for (int r = 0; r < 8; ++r) {
        float v = acc[i][r];
        int ch = chb + r;
        if (MODE == 0) v = gelu_f(v + p0[ch]);
        else           v = v * p0[ch] + p1[ch];
        pk[r] = tb(v);
      }
      *(v8bf*)&oh[(size_t)pixel * CO + chb] = pk;
    }
  }
}

// --------------------- row/col mean pooling (128ch NHWC) -------------------
__global__ __launch_bounds__(128) void rowmean128(const bf16_t* __restrict__ in,
                                                  float* __restrict__ out) {
  int bid = blockIdx.x;                 // b*512 + y
  int y = bid & 511, b = bid >> 9;
  int n = threadIdx.x;
  float acc = 0.f;
  size_t base = (size_t)(b * HH + y) * WW * 128 + n;
  for (int x = 0; x < WW; ++x) acc += fb(in[base + (size_t)x * 128]);
  out[((size_t)b * 128 + n) * HH + y] = acc * (1.f / 512.f);
}

__global__ __launch_bounds__(256) void colmean128(const bf16_t* __restrict__ in,
                                                  float* __restrict__ out) {
  int bid = blockIdx.x;                 // b*256 + xpair
  int b = bid >> 8;
  int x = (bid & 255) * 2 + (threadIdx.x >> 7);
  int n = threadIdx.x & 127;
  float acc = 0.f;
  for (int y = 0; y < HH; ++y)
    acc += fb(in[((size_t)(b * HH + y) * WW + x) * 128 + n]);
  out[((size_t)b * 128 + n) * WW + x] = acc * (1.f / 512.f);
}

// ------------------- axial attention (16x16 per b,head) --------------------
__global__ __launch_bounds__(256) void axial_attn(
    const float* __restrict__ qm, const float* __restrict__ km,
    const float* __restrict__ vm, const float* __restrict__ peq,
    const float* __restrict__ pek, float* __restrict__ out) {
  __shared__ float S[256];
  int b = blockIdx.x >> 3, hd = blockIdx.x & 7;
  int nb = hd * 16;
  int t = threadIdx.x;
  int i = t >> 4, j = t & 15;
  const float* q  = qm  + ((size_t)b * 128 + nb + i) * 512;
  const float* pq = peq + (size_t)(nb + i) * 512;
  const float* k  = km  + ((size_t)b * 128 + nb + j) * 512;
  const float* pk = pek + (size_t)(nb + j) * 512;
  float acc = 0.f;
  for (int e = 0; e < 512; ++e) acc += (q[e] + pq[e]) * (k[e] + pk[e]);
  S[i * 16 + j] = acc * 0.04419417382415922f;   // 1/sqrt(512)
  __syncthreads();
  if (t < 16) {
    float mx = -1e30f;
    for (int jj = 0; jj < 16; ++jj) mx = fmaxf(mx, S[t * 16 + jj]);
    float sum = 0.f; float e16[16];
    for (int jj = 0; jj < 16; ++jj) { float e = expf(S[t * 16 + jj] - mx); e16[jj] = e; sum += e; }
    float r = 1.f / sum;
    for (int jj = 0; jj < 16; ++jj) S[t * 16 + jj] = e16[jj] * r;
  }
  __syncthreads();
  for (int ii = 0; ii < 16; ++ii) {
    #pragma unroll
    for (int half = 0; half < 2; ++half) {
      int e = t + half * 256;
      float a = 0.f;
      #pragma unroll
      for (int jj = 0; jj < 16; ++jj)
        a += S[ii * 16 + jj] * vm[((size_t)b * 128 + nb + jj) * 512 + e];
      out[((size_t)b * 128 + nb + ii) * 512 + e] = a;
    }
  }
}

// avec[b][n][x] = ape_row[n]*attn_row[b,n,x] + ape_col[n]*attn_col[b,n,x]
__global__ __launch_bounds__(256) void avec_kernel(
    const float* __restrict__ ar, const float* __restrict__ ac,
    const float* __restrict__ wr, const float* __restrict__ wc,
    float* __restrict__ av) {
  int idx = blockIdx.x * 256 + threadIdx.x;   // ((b*128+n)*512 + x)
  int n = (idx >> 9) & 127;
  av[idx] = wr[n] * ar[idx] + wc[n] * ac[idx];
}

// ----- cat + channel-shuffle + depthwise 3x3 + bias + GELU  -> (P,320) -----
__global__ __launch_bounds__(320) void tcat_dw(
    const bf16_t* __restrict__ xh, const bf16_t* __restrict__ vmap,
    const float* __restrict__ av, const float* __restrict__ w,
    const float* __restrict__ bias, bf16_t* __restrict__ tout) {
  __shared__ bf16_t cat[3 * 34 * 320];   // unshuffled cat halo tile, 65280B
  int bid  = blockIdx.x;
  int xblk = bid & 15, y = (bid >> 4) & 511, b = bid >> 13;
  int t = threadIdx.x;
  // chans 0..63 : x (bf16 NHWC)
  for (int i = t; i < 3 * 34 * 64; i += 320) {
    int c = i & 63, pr = i >> 6;
    int r = pr / 34, px = pr % 34;
    int yy = y + r - 1, xx = xblk * 32 + px - 1;
    float v = 0.f;
    if (yy >= 0 && yy < HH && xx >= 0 && xx < WW)
      v = fb(xh[((size_t)(b * HH + yy) * WW + xx) * 64 + c]);
    cat[(r * 34 + px) * 320 + c] = tb(v);
  }
  // chans 64..191 : attn = v + avec ; chans 192..319 : v
  for (int i = t; i < 3 * 34 * 128; i += 320) {
    int n = i & 127, pr = i >> 7;
    int r = pr / 34, px = pr % 34;
    int yy = y + r - 1, xx = xblk * 32 + px - 1;
    float vv = 0.f, aa = 0.f;
    if (yy >= 0 && yy < HH && xx >= 0 && xx < WW) {
      vv = fb(vmap[((size_t)(b * HH + yy) * WW + xx) * 128 + n]);
      aa = av[((size_t)b * 128 + n) * WW + xx];
    }
    cat[(r * 34 + px) * 320 + 64 + n]  = tb(vv + aa);
    cat[(r * 34 + px) * 320 + 192 + n] = tb(vv);
  }
  __syncthreads();
  int oc  = t;
  int src = (oc & 63) * 5 + (oc >> 6);   // channel shuffle g=5
  float w9[9];
  #pragma unroll
  for (int k = 0; k < 9; ++k) w9[k] = w[oc * 9 + k];
  float b0 = bias[oc];
  for (int xi = 0; xi < 32; ++xi) {
    float acc = 0.f;
    #pragma unroll
    for (int r = 0; r < 3; ++r)
      #pragma unroll
      for (int dx = 0; dx < 3; ++dx)
        acc += w9[r * 3 + dx] * fb(cat[(r * 34 + xi + dx) * 320 + src]);
    int x = xblk * 32 + xi;
    tout[((size_t)(b * HH + y) * WW + x) * 320 + oc] = tb(gelu_f(acc + b0));
  }
}

// ---------------------------------------------------------------------------
static inline size_t alignup(size_t v) { return (v + 255) & ~(size_t)255; }

extern "C" void kernel_launch(void* const* d_in, const int* in_sizes, int n_in,
                              void* d_out, int out_size, void* d_ws, size_t ws_size,
                              hipStream_t stream) {
  const float* x          = (const float*)d_in[0];
  const float* shortcut   = (const float*)d_in[1];
  const float* ln_w       = (const float*)d_in[2];
  const float* ln_b       = (const float*)d_in[3];
  const float* dproj_dw_w = (const float*)d_in[4];
  const float* dproj_bn_s = (const float*)d_in[5];
  const float* dproj_bn_t = (const float*)d_in[6];
  const float* dproj_pw_w = (const float*)d_in[7];
  const float* dproj_pw_b = (const float*)d_in[8];
  const float* q_dw_w     = (const float*)d_in[9];
  const float* q_bn_s     = (const float*)d_in[10];
  const float* q_bn_t     = (const float*)d_in[11];
  const float* k_dw_w     = (const float*)d_in[12];
  const float* k_bn_s     = (const float*)d_in[13];
  const float* k_bn_t     = (const float*)d_in[14];
  const float* v_pw_w     = (const float*)d_in[15];
  const float* v_bn_s     = (const float*)d_in[16];
  const float* v_bn_t     = (const float*)d_in[17];
  const float* proj_dw_w  = (const float*)d_in[18];
  const float* proj_dw_b  = (const float*)d_in[19];
  const float* proj_pw_w  = (const float*)d_in[20];
  const float* proj_pw_b  = (const float*)d_in[21];
  const float* pe_rowq    = (const float*)d_in[22];
  const float* pe_rowk    = (const float*)d_in[23];
  const float* pe_colq    = (const float*)d_in[24];
  const float* pe_colk    = (const float*)d_in[25];
  const float* ape_row_w  = (const float*)d_in[26];
  const float* ape_col_w  = (const float*)d_in[27];
  float* outf = (float*)d_out;

  char* ws = (char*)d_ws;
  size_t off = 0;
  auto take = [&](size_t bytes) { size_t o = off; off = alignup(off + bytes); return o; };
  bf16_t* xh   = (bf16_t*)(ws + take((size_t)PIX * 64 * 2));
  bf16_t* sn   = (bf16_t*)(ws + take((size_t)PIX * 64 * 2));
  bf16_t* s1   = (bf16_t*)(ws + take((size_t)PIX * 64 * 2));
  bf16_t* s    = (bf16_t*)(ws + take((size_t)PIX * 128 * 2));
  bf16_t* qm   = (bf16_t*)(ws + take((size_t)PIX * 128 * 2));
  bf16_t* km   = (bf16_t*)(ws + take((size_t)PIX * 128 * 2));
  bf16_t* vmap = (bf16_t*)(ws + take((size_t)PIX * 128 * 2));
  bf16_t* tbuf = (bf16_t*)(ws + take((size_t)PIX * 320 * 2));
  bf16_t* wbd  = (bf16_t*)(ws + take(128 * 64 * 2));
  bf16_t* wbv  = (bf16_t*)(ws + take(128 * 64 * 2));
  bf16_t* wbp  = (bf16_t*)(ws + take(64 * 320 * 2));
  const size_t VEC = (size_t)BB * 128 * 512 * 4;   // (B,128,512) f32
  float* qrow = (float*)(ws + take(VEC));
  float* qcol = (float*)(ws + take(VEC));
  float* krow = (float*)(ws + take(VEC));
  float* kcol = (float*)(ws + take(VEC));
  float* vrow = (float*)(ws + take(VEC));
  float* vcol = (float*)(ws + take(VEC));
  float* arow = (float*)(ws + take(VEC));
  float* acol = (float*)(ws + take(VEC));
  float* avec = (float*)(ws + take(VEC));

  // 1. weights -> bf16
  wconv_kernel<<<80, 256, 0, stream>>>(dproj_pw_w, v_pw_w, proj_pw_w, wbd, wbv, wbp);
  // 2. x -> NHWC bf16 ; LayerNorm(shortcut) -> NHWC bf16
  nchw2nhwc64<<<PIX / 256, 256, 0, stream>>>(x, xh);
  ln_kernel<<<PIX / 256, 256, 0, stream>>>(shortcut, ln_w, ln_b, sn);
  // 3. dproj: dw3x3+BN+GELU, then pw 64->128 (+bias,+GELU) via WMMA
  dw3x3_bn_gelu<64, 0><<<16384, 256, 0, stream>>>(sn, 64, 0, dproj_dw_w, dproj_bn_s, dproj_bn_t, s1);
  gemm_pw<64, 128, 64, 0><<<PIX / 128, 256, 0, stream>>>(s1, 0, wbd, dproj_pw_b, nullptr, s, nullptr);
  // 4. q/k grouped dw3x3+BN+GELU ; v pw 64->128 + BN via WMMA
  dw3x3_bn_gelu<128, 1><<<16384, 256, 0, stream>>>(xh, 64, 0, q_dw_w, q_bn_s, q_bn_t, qm);
  dw3x3_bn_gelu<128, 1><<<16384, 256, 0, stream>>>(s, 128, 0, k_dw_w, k_bn_s, k_bn_t, km);
  gemm_pw<64, 128, 128, 1><<<PIX / 128, 256, 0, stream>>>(s, 64, wbv, v_bn_s, v_bn_t, vmap, nullptr);
  // 5. row/col mean pooling
  rowmean128<<<BB * HH, 128, 0, stream>>>(qm, qrow);
  rowmean128<<<BB * HH, 128, 0, stream>>>(km, krow);
  rowmean128<<<BB * HH, 128, 0, stream>>>(vmap, vrow);
  colmean128<<<BB * 256, 256, 0, stream>>>(qm, qcol);
  colmean128<<<BB * 256, 256, 0, stream>>>(km, kcol);
  colmean128<<<BB * 256, 256, 0, stream>>>(vmap, vcol);
  // 6. axial attention (row + col), then fold to per-(b,n,x) vector
  axial_attn<<<16, 256, 0, stream>>>(qrow, krow, vrow, pe_rowq, pe_rowk, arow);
  axial_attn<<<16, 256, 0, stream>>>(qcol, kcol, vcol, pe_colq, pe_colk, acol);
  avec_kernel<<<(BB * 128 * 512) / 256, 256, 0, stream>>>(arow, acol, ape_row_w, ape_col_w, avec);
  // 7. cat + shuffle + dw3x3 + GELU -> (P,320) bf16
  tcat_dw<<<16384, 320, 0, stream>>>(xh, vmap, avec, proj_dw_w, proj_dw_b, tbuf);
  // 8. final pw 320->64 + bias + GELU -> f32 NCHW output (WMMA)
  gemm_pw<320, 64, 320, 2><<<PIX / 128, 256, 0, stream>>>(tbuf, 0, wbp, proj_pw_b, nullptr, nullptr, outf);
  (void)n_in; (void)in_sizes; (void)out_size; (void)ws_size;
}
